// WindowAttention_214748365481
// MI455X (gfx1250) — matrix operand order repaired and checked by reading, not compiled, not run on hardware
//
#include <hip/hip_runtime.h>
#include <hip/hip_bf16.h>
#include <math.h>

// ---------------------------------------------------------------------------
// 3D shifted-window attention for MI455X (gfx1250, wave32, WMMA f16 16x16x32)
//
// Roofline: ~69 GFLOP, >=250 MB HBM traffic. f32 VALU would be compute bound;
// f16 WMMA (f32 accum) pushes compute below the ~15us memory floor, so all
// GEMMs run through v_wmma_f32_16x16x32_f16. Bias table (1MB) precomputed,
// L2-resident across all 1024 attention combos. Flash-style online softmax
// keeps the 512x512 score matrix out of memory entirely.
// ---------------------------------------------------------------------------

typedef _Float16 f16;
typedef __attribute__((ext_vector_type(16))) _Float16 v16h;
typedef __attribute__((ext_vector_type(8)))  _Float16 v8h;
typedef __attribute__((ext_vector_type(8)))  float    v8f;

#define HEADS 8
#define HD    32
#define DIM   256          // HEADS*HD
#define T     512          // tokens per window (8^3)
#define NWIN  64           // windows per batch (4^3)
#define NB    2

__device__ __forceinline__ v8h ld8(const f16* p) { return *(const v8h*)p; }

__device__ __forceinline__ v16h mk16(v8h lo, v8h hi) {
  v16h r;
#pragma unroll
  for (int i = 0; i < 8; ++i) { r[i] = lo[i]; r[i + 8] = hi[i]; }
  return r;
}

__device__ __forceinline__ v8f wmma16(v16h a, v16h b, v8f c) {
  // D = A(16x32 f16) * B(32x16 f16) + C(16x16 f32)
  return __builtin_amdgcn_wmma_f32_16x16x32_f16(false, a, false, b,
                                                (short)0, c, false, false);
}

// ---------------------------------------------------------------------------
// Kernel 0: bias table (512x512 f32) + weight transpose/convert to f16.
//   bias[i][j] = pos[jx-ix+7][jy-iy+7][jz-iz+7]
//   wqkvT[n][k] = (f16) w_qkv[k][n]   (768x256)
//   woutT[n][k] = (f16) w_out[k][n]   (256x256)
// ---------------------------------------------------------------------------
__global__ void prep_kernel(const float* __restrict__ w_qkv,
                            const float* __restrict__ pos,
                            const float* __restrict__ w_out,
                            float* __restrict__ bias,
                            f16* __restrict__ wqkvT,
                            f16* __restrict__ woutT) {
  int idx = blockIdx.x * 256 + threadIdx.x;
  if (idx < 512 * 512) {
    int i = idx >> 9, j = idx & 511;
    int ix = i >> 6, iy = (i >> 3) & 7, iz = i & 7;
    int jx = j >> 6, jy = (j >> 3) & 7, jz = j & 7;
    int rx = jx - ix + 7, ry = jy - iy + 7, rz = jz - iz + 7;
    bias[idx] = pos[(rx * 15 + ry) * 15 + rz];
  } else if (idx < 512 * 512 + 768 * 256) {
    int k2 = idx - 512 * 512;
    int n = k2 >> 8, k = k2 & 255;
    wqkvT[k2] = (f16)w_qkv[k * 768 + n];
  } else {
    int k2 = idx - 512 * 512 - 768 * 256;   // < 256*256 by grid size
    int n = k2 >> 8, k = k2 & 255;
    woutT[k2] = (f16)w_out[k * 256 + n];
  }
}

// ---------------------------------------------------------------------------
// Kernel 1: QKV projection with fused roll(-4) + window gather.
// One block = one (b, window, 16-token tile). 16 rows of x staged in LDS as
// f16, then 8 waves x 6 n-tiles of WMMA over K=256. Outputs:
//   qbuf/kbuf : [b][h][win][t][d]   f16 (row-major, 32-f16 rows)
//   vT        : [b][h][win][d][t]   f16 (transposed so P*V B-operand loads
//                                        are contiguous 32B per lane)
// ---------------------------------------------------------------------------
__global__ void qkv_kernel(const float* __restrict__ x,
                           const f16* __restrict__ wqkvT,
                           f16* __restrict__ qbuf,
                           f16* __restrict__ kbuf,
                           f16* __restrict__ vT) {
  __shared__ __align__(16) f16 xs[16][DIM];
  int bid = blockIdx.x;
  int tt = bid & 31;
  int w  = (bid >> 5) & 63;
  int b  = bid >> 11;
  int wx = w >> 4, wy = (w >> 2) & 3, wz = w & 3;
  int t0 = tt * 16;
  int tid = threadIdx.x;

  // Stage 16 rows (rolled gather) as f16
  {
    int row = tid >> 4;
    int cs  = (tid & 15) * 16;
    int t = t0 + row;
    int tx = t >> 6, ty = (t >> 3) & 7, tz = t & 7;
    int ox = (wx * 8 + tx + 4) & 31;
    int oy = (wy * 8 + ty + 4) & 31;
    int oz = (wz * 8 + tz + 4) & 31;
    const float* xr = x + ((((size_t)b * 32 + ox) * 32 + oy) * 32 + oz) * DIM + cs;
#pragma unroll
    for (int i = 0; i < 16; ++i) xs[row][cs + i] = (f16)xr[i];
  }
  __syncthreads();

  int lane = tid & 31, wv = tid >> 5;
  int col = lane & 15, half = lane >> 4;

  for (int ni = 0; ni < 6; ++ni) {
    int n0 = (wv * 6 + ni) * 16;          // 0..752
    v8f acc = {};
#pragma unroll
    for (int kc = 0; kc < 8; ++kc) {
      int kb = kc * 32;
      v16h A  = mk16(ld8(&xs[col][kb + half * 16]),
                     ld8(&xs[col][kb + half * 16 + 8]));
      const f16* br = wqkvT + (size_t)(n0 + col) * 256 + kb + half * 16;
      v16h Bm = mk16(ld8(br), ld8(br + 8));
      acc = wmma16(A, Bm, acc);
    }
    int sel = n0 >> 8;              // 0=q 1=k 2=v
    int h   = (n0 >> 5) & 7;
    int dq  = n0 & 31;              // 0 or 16
    size_t whw = ((size_t)b * HEADS + h) * NWIN + w;
    if (sel < 2) {
      f16* dst = (sel == 0 ? qbuf : kbuf) + whw * T * HD;
#pragma unroll
      for (int r = 0; r < 8; ++r) {
        int t = t0 + r + 8 * half;
        dst[(size_t)t * HD + dq + col] = (f16)acc[r];
      }
    } else {
      f16* dst = vT + whw * HD * T;
#pragma unroll
      for (int r = 0; r < 8; ++r) {
        int t = t0 + r + 8 * half;
        dst[(size_t)(dq + col) * T + t] = (f16)acc[r];
      }
    }
  }
}

// ---------------------------------------------------------------------------
// Kernel 2: flash attention per (b, head, window, 16-row chunk). One wave per
// chunk; 8 waves per block. Scores: 32x v_wmma (16 keys each), online softmax
// with half-wave shfl reductions; probs re-laid C->A through 1KB/wave LDS;
// P*V: 2 WMMAs per 32-key pair. Output f16 to attnout[b][win][t][h*32+d].
// ---------------------------------------------------------------------------
__global__ void attn_kernel(const f16* __restrict__ qbuf,
                            const f16* __restrict__ kbuf,
                            const f16* __restrict__ vT,
                            const float* __restrict__ bias,
                            f16* __restrict__ attnout) {
  __shared__ __align__(16) f16 probs[8][16][32];
  int tid = threadIdx.x;
  int lane = tid & 31, wv = tid >> 5;
  int gid = blockIdx.x * 8 + wv;          // 32768 waves total
  int chunk = gid & 31;
  int combo = gid >> 5;
  int w = combo & 63;
  int h = (combo >> 6) & 7;
  int b = combo >> 9;
  int i0 = chunk * 16;
  int col = lane & 15, half = lane >> 4;
  const float scale = 0.17677669529663687f;   // 1/sqrt(32)

  size_t whw = ((size_t)b * HEADS + h) * NWIN + w;
  const f16* qb = qbuf + whw * T * HD;
  const f16* kb = kbuf + whw * T * HD;
  const f16* vb = vT   + whw * HD * T;

  // q tile in A layout: row = i0+col, K range half*16..+15
  const f16* qr = qb + (size_t)(i0 + col) * HD + half * 16;
  v16h qa = mk16(ld8(qr), ld8(qr + 8));

  float m[8], l[8];
  v8f o0 = {}, o1 = {};
#pragma unroll
  for (int r = 0; r < 8; ++r) { m[r] = -1e30f; l[r] = 0.f; }

  for (int j = 0; j < 32; ++j) {
    int jb = j * 16;
    // B = k^T : lane col -> key jb+col, K(=d) range half*16..+15
    const f16* kr = kb + (size_t)(jb + col) * HD + half * 16;
    v16h kB = mk16(ld8(kr), ld8(kr + 8));
    v8f zc = {};
    v8f s = wmma16(qa, kB, zc);

    const float* biasp = bias + (size_t)(i0 + 8 * half) * 512 + jb + col;
#pragma unroll
    for (int r = 0; r < 8; ++r) {
      float sv = s[r] * scale + biasp[(size_t)r * 512];
      float tm = sv;
      tm = fmaxf(tm, __shfl_xor(tm, 8, 16));
      tm = fmaxf(tm, __shfl_xor(tm, 4, 16));
      tm = fmaxf(tm, __shfl_xor(tm, 2, 16));
      tm = fmaxf(tm, __shfl_xor(tm, 1, 16));
      float mn = fmaxf(m[r], tm);
      float fs = __expf(m[r] - mn);
      float pv = __expf(sv - mn);
      float ps = pv;
      ps += __shfl_xor(ps, 8, 16);
      ps += __shfl_xor(ps, 4, 16);
      ps += __shfl_xor(ps, 2, 16);
      ps += __shfl_xor(ps, 1, 16);
      l[r] = l[r] * fs + ps;
      m[r] = mn;
      o0[r] *= fs;
      o1[r] *= fs;
      probs[wv][r + 8 * half][(j & 1) * 16 + col] = (f16)pv;
    }

    if (j & 1) {
      // wave-private LDS: wait for our ds stores before re-reading in A layout
      asm volatile("s_wait_dscnt 0" ::: "memory");
      v16h pA = mk16(ld8(&probs[wv][col][half * 16]),
                     ld8(&probs[wv][col][half * 16 + 8]));
      int jb2 = (j - 1) * 16;               // 32-key chunk base
      const f16* vr0 = vb + (size_t)col * T        + jb2 + half * 16;  // d=col
      const f16* vr1 = vb + (size_t)(col + 16) * T + jb2 + half * 16;  // d=col+16
      v16h vB0 = mk16(ld8(vr0), ld8(vr0 + 8));
      v16h vB1 = mk16(ld8(vr1), ld8(vr1 + 8));
      o0 = wmma16(pA, vB0, o0);
      o1 = wmma16(pA, vB1, o1);
    }
  }

#pragma unroll
  for (int r = 0; r < 8; ++r) {
    float inv = 1.0f / l[r];
    o0[r] *= inv;
    o1[r] *= inv;
  }
  f16* ao = attnout + (((size_t)b * NWIN + w) * T) * DIM + h * HD;
#pragma unroll
  for (int r = 0; r < 8; ++r) {
    int t = i0 + r + 8 * half;
    ao[(size_t)t * DIM + col]      = (f16)o0[r];
    ao[(size_t)t * DIM + 16 + col] = (f16)o1[r];
  }
}

// ---------------------------------------------------------------------------
// Kernel 3: output projection (256x256) + b_out + roll(+4) scatter to d_out.
// ---------------------------------------------------------------------------
__global__ void proj_kernel(const f16* __restrict__ attnout,
                            const f16* __restrict__ woutT,
                            const float* __restrict__ b_out,
                            float* __restrict__ out) {
  __shared__ __align__(16) f16 as_[16][DIM];
  int bid = blockIdx.x;
  int tt = bid & 31;
  int w  = (bid >> 5) & 63;
  int b  = bid >> 11;
  int wx = w >> 4, wy = (w >> 2) & 3, wz = w & 3;
  int t0 = tt * 16;
  int tid = threadIdx.x;
  {
    int row = tid >> 4;
    int cs  = (tid & 15) * 16;
    const f16* ar = attnout + (((size_t)b * NWIN + w) * T + t0 + row) * DIM + cs;
    *(v8h*)&as_[row][cs]     = ld8(ar);
    *(v8h*)&as_[row][cs + 8] = ld8(ar + 8);
  }
  __syncthreads();

  int lane = tid & 31, wv = tid >> 5;
  int col = lane & 15, half = lane >> 4;
  for (int ni = 0; ni < 2; ++ni) {
    int n0 = (wv * 2 + ni) * 16;
    v8f acc = {};
#pragma unroll
    for (int kc = 0; kc < 8; ++kc) {
      int kb = kc * 32;
      v16h A  = mk16(ld8(&as_[col][kb + half * 16]),
                     ld8(&as_[col][kb + half * 16 + 8]));
      const f16* br = woutT + (size_t)(n0 + col) * 256 + kb + half * 16;
      v16h Bm = mk16(ld8(br), ld8(br + 8));
      acc = wmma16(A, Bm, acc);
    }
    float bo = b_out[n0 + col];
#pragma unroll
    for (int r = 0; r < 8; ++r) {
      int t = t0 + r + 8 * half;
      int tx = t >> 6, ty = (t >> 3) & 7, tz = t & 7;
      int ox = (wx * 8 + tx + 4) & 31;
      int oy = (wy * 8 + ty + 4) & 31;
      int oz = (wz * 8 + tz + 4) & 31;
      out[((((size_t)b * 32 + ox) * 32 + oy) * 32 + oz) * DIM + n0 + col] =
          acc[r] + bo;
    }
  }
}

// ---------------------------------------------------------------------------
extern "C" void kernel_launch(void* const* d_in, const int* in_sizes, int n_in,
                              void* d_out, int out_size, void* d_ws,
                              size_t ws_size, hipStream_t stream) {
  const float* x    = (const float*)d_in[0];
  const float* wqkv = (const float*)d_in[1];
  const float* pos  = (const float*)d_in[2];
  const float* wout = (const float*)d_in[3];
  const float* bout = (const float*)d_in[4];

  char* ws = (char*)d_ws;
  const size_t BIG = 33554432;                 // 32MB per f16 qkv buffer
  float* bias  = (float*)(ws);                              // 1 MB
  f16* wqkvT   = (f16*)(ws + 1048576);                      // 384 KB
  f16* woutT   = (f16*)(ws + 1441792);                      // 128 KB
  f16* qbuf    = (f16*)(ws + 1572864);                      // 32 MB
  f16* kbuf    = (f16*)(ws + 1572864 + BIG);                // 32 MB
  f16* vT      = (f16*)(ws + 1572864 + 2 * BIG);            // 32 MB
  f16* attnout = (f16*)(ws + 1572864 + 3 * BIG);            // 32 MB

  prep_kernel<<<2048, 256, 0, stream>>>(wqkv, pos, wout, bias, wqkvT, woutT);
  qkv_kernel<<<4096, 256, 0, stream>>>(x, wqkvT, qbuf, kbuf, vT);
  attn_kernel<<<4096, 256, 0, stream>>>(qbuf, kbuf, vT, bias, attnout);
  proj_kernel<<<4096, 256, 0, stream>>>(attnout, woutT, bout, (float*)d_out);
}
